// DifferentialAttention_58978490908694
// MI455X (gfx1250) — matrix operand order repaired and checked by reading, not compiled, not run on hardware
//
#include <hip/hip_runtime.h>
#include <hip/hip_bf16.h>

typedef __bf16 bf16;
typedef __attribute__((ext_vector_type(16))) __bf16        v16bf;
typedef __attribute__((ext_vector_type(8)))  float         v8f;
typedef __attribute__((ext_vector_type(4)))  int           v4i;
typedef __attribute__((ext_vector_type(8)))  int           v8i;
typedef __attribute__((ext_vector_type(4)))  unsigned int  v4u;

union ABFrag { v16bf v; v4i i2[2]; };

#define BATCH     2
#define S_LEN     2048
#define EMB       1024
#define NUM_HEADS 8
#define HEAD_DIM  64
#define M_ROWS    (BATCH * S_LEN)   // 4096
#define NQKV      1536              // 1024 q + 256 k + 256 v
#define LAMBDA_INIT 0.8f
#define EPS_RMS   1e-5f

#if defined(__has_builtin)
#  if __has_builtin(__builtin_amdgcn_tensor_load_to_lds) && __has_builtin(__builtin_amdgcn_s_wait_tensorcnt)
#    define HAVE_TDM 1
#  endif
#endif

// ---------------------------------------------------------------------------
// Fragment loaders per CDNA5 ISA 7.12.2 (wave32, 16-bit operands).
// A 16x32: lane r=l&15 is row; VGPR0..3 = K {half*8..+7}, VGPR4..7 = {16+half*8..+7}
// B 32x16: lane n=l&15 is col; VGPR0..3 = K {half*8..+7}, VGPR4..7 = {16+half*8..+7}
// Two 16-byte loads each when the source row is contiguous in K.
// ---------------------------------------------------------------------------
static __device__ inline v16bf loadFragA(const bf16* row, int kb, int half) {
    ABFrag f;
    f.i2[0] = *(const v4i*)(row + kb + half * 8);
    f.i2[1] = *(const v4i*)(row + kb + 16 + half * 8);
    return f.v;
}
static __device__ inline v16bf loadFragB(const bf16* rowT, int kb, int half) {
    ABFrag f;
    f.i2[0] = *(const v4i*)(rowT + kb + half * 8);
    f.i2[1] = *(const v4i*)(rowT + kb + 16 + half * 8);
    return f.v;
}

// ---------------------------------------------------------------------------
// Tensor Data Mover: DMA a 2-D bf16 tile (rows x rowElems, row stride in
// elements) from global into LDS. Descriptor bit layout per ISA 08 §8.3/8.4:
//   G0: count=1 | lds_addr | global_addr[56:0] | type=2
//   G1: data_size=1(2B) | tensor_dim0/1 | tile_dim0/1 | tensor_dim0_stride
// Issued once by one wave (EXEC-independent), completion via TENSORcnt.
// ---------------------------------------------------------------------------
static __device__ inline void tdm_load_2d(const bf16* gsrc, bf16* lds_dst,
                                          int rows, int rowElems, int rowStrideElems) {
#if defined(HAVE_TDM)
    unsigned long long ga = (unsigned long long)(size_t)gsrc;
    unsigned int       la = (unsigned int)(size_t)lds_dst;  // LDS aperture: addr[31:0]
    v4u g0;
    g0[0] = 1u;                                              // count=1, user mode
    g0[1] = la;                                              // lds_addr
    g0[2] = (unsigned int)(ga & 0xffffffffull);              // global_addr[31:0]
    g0[3] = (unsigned int)((ga >> 32) & 0x1ffffffull) | 0x80000000u; // [56:32] | type=2
    v8i g1;
    g1[0] = (1 << 16);                                       // data_size = 2 bytes
    g1[1] = (rowElems & 0xffff) << 16;                       // tensor_dim0 lo
    g1[2] = ((rowElems >> 16) & 0xffff) | ((rows & 0xffff) << 16);     // dim0 hi | dim1 lo
    g1[3] = ((rows >> 16) & 0xffff) | ((rowElems & 0xffff) << 16);     // dim1 hi | tile_dim0
    g1[4] = rows & 0xffff;                                   // tile_dim1 (tile_dim2 = 0)
    g1[5] = rowStrideElems;                                  // tensor_dim0_stride[31:0]
    g1[6] = 0;                                               // stride hi / dim1_stride lo
    g1[7] = 0;
    v4i gz = {0, 0, 0, 0};
#  if defined(__clang_major__) && (__clang_major__ >= 23)
    v8i gz8 = {0, 0, 0, 0, 0, 0, 0, 0};
    __builtin_amdgcn_tensor_load_to_lds(g0, g1, gz, gz, gz8, 0);
#  else
    __builtin_amdgcn_tensor_load_to_lds(g0, g1, gz, gz, 0);
#  endif
#endif
}

// ---------------------------------------------------------------------------
// bf16 GEMM, double-buffered TDM pipeline:
//   C[M,N] = A[M,K] (row-major) * BT[N,K]^T
// Block = 256 threads = 8 waves (4 along M x 2 along N); block tile 128x128,
// K consumed in 64-deep chunks. Wave 0 issues the NEXT chunk's two
// tensor_load_to_lds into the ping-pong buffer, then waits tensorcnt<=2:
// same-wave tensor ops complete in order, so the CURRENT chunk's pair is
// guaranteed resident while the next chunk's DMA overlaps 16 WMMAs/wave.
// LDS: 2 bufs x (A 128x64 + B 128x64) bf16 = 64 KB.
// ---------------------------------------------------------------------------
#define KCH 64
__global__ __launch_bounds__(256)
void gemm_bf16(const bf16* __restrict__ A, const bf16* __restrict__ BT,
               float* __restrict__ C, int M, int N, int K, int ldc) {
    __shared__ __align__(16) bf16 Atile[2][128 * KCH];   // 2 x 16 KB
    __shared__ __align__(16) bf16 Btile[2][128 * KCH];   // 2 x 16 KB

    const int lane = threadIdx.x & 31;
    const int wid  = threadIdx.x >> 5;
    const int wm   = wid & 3;              // wave M slot: 0..3  (32 rows each)
    const int wn   = wid >> 2;             // wave N slot: 0..1  (64 cols each)
    const int mBlk = blockIdx.y * 128;
    const int nBlk = blockIdx.x * 128;
    const int r = lane & 15, half = lane >> 4;
    const int nIter = K / KCH;

    v8f acc[2][4];
    const v8f vz = {0.f, 0.f, 0.f, 0.f, 0.f, 0.f, 0.f, 0.f};
#pragma unroll
    for (int i = 0; i < 2; ++i)
#pragma unroll
        for (int t = 0; t < 4; ++t) acc[i][t] = vz;

#if defined(HAVE_TDM)
    if (wid == 0) {   // prologue: chunk 0 into buffer 0
        tdm_load_2d(A  + (size_t)mBlk * K, &Atile[0][0], 128, KCH, K);
        tdm_load_2d(BT + (size_t)nBlk * K, &Btile[0][0], 128, KCH, K);
    }
#endif

    for (int it = 0; it < nIter; ++it) {
        const int buf = it & 1;
#if defined(HAVE_TDM)
        if (wid == 0) {
            if (it + 1 < nIter) {   // prefetch next chunk into the other buffer
                tdm_load_2d(A  + (size_t)mBlk * K + (it + 1) * KCH, &Atile[buf ^ 1][0], 128, KCH, K);
                tdm_load_2d(BT + (size_t)nBlk * K + (it + 1) * KCH, &Btile[buf ^ 1][0], 128, KCH, K);
                __builtin_amdgcn_s_wait_tensorcnt(2);   // current chunk's pair has landed
            } else {
                __builtin_amdgcn_s_wait_tensorcnt(0);
            }
        }
#else
        // Cooperative fallback copy (toolchains without the TDM builtin)
        for (int idx = threadIdx.x; idx < 128 * KCH / 8; idx += 256) {
            int e = idx * 8, row = e / KCH, col = e % KCH;
            *(v4i*)(&Atile[buf][0] + e) = *(const v4i*)(A  + ((size_t)mBlk + row) * K + it * KCH + col);
            *(v4i*)(&Btile[buf][0] + e) = *(const v4i*)(BT + ((size_t)nBlk + row) * K + it * KCH + col);
        }
#endif
        __syncthreads();

#pragma unroll
        for (int kk = 0; kk < KCH; kk += 32) {
            v16bf a0 = loadFragA(&Atile[buf][0] + (wm * 32 + r) * KCH, kk, half);
            v16bf a1 = loadFragA(&Atile[buf][0] + (wm * 32 + 16 + r) * KCH, kk, half);
#pragma unroll
            for (int t = 0; t < 4; ++t) {
                v16bf b = loadFragB(&Btile[buf][0] + (wn * 64 + t * 16 + r) * KCH, kk, half);
                acc[0][t] = __builtin_amdgcn_wmma_f32_16x16x32_bf16(false, a0, false, b,
                                                                    (short)0, acc[0][t], false, false);
                acc[1][t] = __builtin_amdgcn_wmma_f32_16x16x32_bf16(false, a1, false, b,
                                                                    (short)0, acc[1][t], false, false);
            }
        }
        __syncthreads();   // all waves done with buf before its next DMA is issued
    }

    // C layout: VGPR i -> row +i + 8*half, lanes -> n
#pragma unroll
    for (int i = 0; i < 2; ++i) {
        const int m0 = mBlk + wm * 32 + i * 16 + half * 8;
#pragma unroll
        for (int t = 0; t < 4; ++t) {
            const int n0 = nBlk + wn * 64 + t * 16 + r;
            float* crow = C + (size_t)m0 * ldc + n0;
#pragma unroll
            for (int j = 0; j < 8; ++j) crow[(size_t)j * ldc] = acc[i][t][j];
        }
    }
}

// ---------------------------------------------------------------------------
// Conversion / packing kernels
// ---------------------------------------------------------------------------
__global__ void cvt_x(const float* __restrict__ x, bf16* __restrict__ xb, int n) {
    int i = blockIdx.x * 256 + threadIdx.x;
    if (i < n) xb[i] = (bf16)x[i];
}

__global__ void pack_wqkvT(const float* __restrict__ Wq, const float* __restrict__ Wk,
                           const float* __restrict__ Wv, bf16* __restrict__ wT) {
    int i = blockIdx.x * 256 + threadIdx.x;
    if (i >= NQKV * EMB) return;
    int n = i / EMB, k = i % EMB;
    float v;
    if (n < 1024)      v = Wq[(size_t)k * 1024 + n];
    else if (n < 1280) v = Wk[(size_t)k * 256 + (n - 1024)];
    else               v = Wv[(size_t)k * 256 + (n - 1280)];
    wT[(size_t)n * EMB + k] = (bf16)v;
}

__global__ void pack_woT(const float* __restrict__ Wo, bf16* __restrict__ wT) {
    int i = blockIdx.x * 256 + threadIdx.x;
    if (i >= EMB * EMB) return;
    int n = i / EMB, k = i % EMB;
    wT[(size_t)n * EMB + k] = (bf16)Wo[(size_t)k * EMB + n];
}

__global__ void lambda_kernel(const float* __restrict__ lq1, const float* __restrict__ lk1,
                              const float* __restrict__ lq2, const float* __restrict__ lk2,
                              float* __restrict__ lam) {
    __shared__ float s1[64], s2[64];
    int t = threadIdx.x;
    s1[t] = lq1[t] * lk1[t];
    s2[t] = lq2[t] * lk2[t];
    __syncthreads();
    if (t == 0) {
        float a = 0.f, b = 0.f;
        for (int i = 0; i < 64; ++i) { a += s1[i]; b += s2[i]; }
        *lam = expf(a) - expf(b) + LAMBDA_INIT;
    }
}

// ---------------------------------------------------------------------------
// RoPE + repack: qkv f32 [M,1536] ->
//   qb [b][16][S][64] bf16 (RoPE, 1/sqrt(64) folded), kb [b][4][S][64] bf16,
//   vT [b][2][128][S] bf16 (transposed so P*V B-fragments load contiguously)
// ---------------------------------------------------------------------------
__global__ __launch_bounds__(256)
void rope_prep(const float* __restrict__ qkv, const float* __restrict__ cosT,
               const float* __restrict__ sinT, bf16* __restrict__ qb,
               bf16* __restrict__ kb, bf16* __restrict__ vT) {
    const int m = blockIdx.x;
    const int b = m >> 11, s = m & (S_LEN - 1);
    const int t = threadIdx.x;
    const float* row = qkv + (size_t)m * NQKV;

    for (int p = t; p < 512; p += 256) {
        int hh = p >> 5, i = p & 31;
        float t1 = row[hh * 64 + 2 * i], t2 = row[hh * 64 + 2 * i + 1];
        float c = cosT[s * 32 + i], sn = sinT[s * 32 + i];
        size_t o = ((size_t)(b * 16 + hh) * S_LEN + s) * 64 + 2 * i;
        qb[o]     = (bf16)((t1 * c - t2 * sn) * 0.125f);
        qb[o + 1] = (bf16)((t1 * sn + t2 * c) * 0.125f);
    }
    for (int p = t; p < 128; p += 256) {
        int kh = p >> 5, i = p & 31;
        float t1 = row[1024 + kh * 64 + 2 * i], t2 = row[1024 + kh * 64 + 2 * i + 1];
        float c = cosT[s * 32 + i], sn = sinT[s * 32 + i];
        size_t o = ((size_t)(b * 4 + kh) * S_LEN + s) * 64 + 2 * i;
        kb[o]     = (bf16)(t1 * c - t2 * sn);
        kb[o + 1] = (bf16)(t1 * sn + t2 * c);
    }
    if (t < 256) {
        int vh = t >> 7, d = t & 127;
        vT[((size_t)(b * 2 + vh) * 128 + d) * S_LEN + s] = (bf16)row[1280 + t];
    }
}

// ---------------------------------------------------------------------------
// Differential flash attention — WMMA QK^T and P*V, shuffle-based online
// softmax, LDS C->A relayout, O1 - lam*O2, RMS norm.
// ---------------------------------------------------------------------------
__global__ __launch_bounds__(64)
void diff_attn(const bf16* __restrict__ qb, const bf16* __restrict__ kbuf,
               const bf16* __restrict__ vT, const float* __restrict__ lamPtr,
               const float* __restrict__ sub_w, bf16* __restrict__ ao) {
    __shared__ __align__(16) bf16  pLds[2][16 * 32];
    __shared__ __align__(16) float oLds[16 * 128];

    const int bid = blockIdx.x;
    const int qt = bid & 127;
    const int h  = (bid >> 7) & 7;
    const int b  = bid >> 10;
    const int lane = threadIdx.x & 31;
    const int wave = threadIdx.x >> 5;
    const int q0 = qt * 16;
    const int qh = 2 * h + wave;
    const int kh = qh & 3;
    const int vh = h & 1;
    const int r = lane & 15, half = lane >> 4;

    const bf16* qrow = qb + ((size_t)(b * 16 + qh) * S_LEN + q0 + r) * 64;
    const v16bf qa0 = loadFragA(qrow, 0, half);
    const v16bf qa1 = loadFragA(qrow, 32, half);

    const bf16* kbase = kbuf + (size_t)(b * 4 + kh) * S_LEN * 64;
    const bf16* vbase = vT + (size_t)(b * 2 + vh) * 128 * S_LEN;

    v8f O[8];
    const v8f vz = {0.f, 0.f, 0.f, 0.f, 0.f, 0.f, 0.f, 0.f};
#pragma unroll
    for (int t = 0; t < 8; ++t) O[t] = vz;
    float mrow[8], lrow[8];
#pragma unroll
    for (int i = 0; i < 8; ++i) { mrow[i] = -1e30f; lrow[i] = 0.f; }

    bf16* pw = &pLds[wave][0];

    for (int c0 = 0; c0 < S_LEN; c0 += 32) {
        v8f s0 = vz, s1 = vz;
        {
            const bf16* kr0 = kbase + (size_t)(c0 + r) * 64;
            const bf16* kr1 = kbase + (size_t)(c0 + 16 + r) * 64;
            v16bf b00 = loadFragB(kr0, 0, half);
            v16bf b01 = loadFragB(kr0, 32, half);
            v16bf b10 = loadFragB(kr1, 0, half);
            v16bf b11 = loadFragB(kr1, 32, half);
            s0 = __builtin_amdgcn_wmma_f32_16x16x32_bf16(false, qa0, false, b00, (short)0, s0, false, false);
            s0 = __builtin_amdgcn_wmma_f32_16x16x32_bf16(false, qa1, false, b01, (short)0, s0, false, false);
            s1 = __builtin_amdgcn_wmma_f32_16x16x32_bf16(false, qa0, false, b10, (short)0, s1, false, false);
            s1 = __builtin_amdgcn_wmma_f32_16x16x32_bf16(false, qa1, false, b11, (short)0, s1, false, false);
        }
        float alpha[8];
#pragma unroll
        for (int i = 0; i < 8; ++i) {
            float mx = fmaxf(s0[i], s1[i]);
            for (int d = 1; d < 16; d <<= 1) mx = fmaxf(mx, __shfl_xor(mx, d, 32));
            float nm = fmaxf(mrow[i], mx);
            float p0 = expf(s0[i] - nm);
            float p1 = expf(s1[i] - nm);
            s0[i] = p0; s1[i] = p1;
            float sm = p0 + p1;
            for (int d = 1; d < 16; d <<= 1) sm += __shfl_xor(sm, d, 32);
            alpha[i] = expf(mrow[i] - nm);
            lrow[i] = lrow[i] * alpha[i] + sm;
            mrow[i] = nm;
        }
#pragma unroll
        for (int t = 0; t < 8; ++t)
#pragma unroll
            for (int i = 0; i < 8; ++i) O[t][i] *= alpha[i];

#pragma unroll
        for (int i = 0; i < 8; ++i) {
            int m = i + 8 * half;
            pw[m * 32 + r]      = (bf16)s0[i];
            pw[m * 32 + 16 + r] = (bf16)s1[i];
        }
        ABFrag pf;   // same-wave DS ops are in-order
        pf.i2[0] = *(const v4i*)(pw + r * 32 + half * 8);
        pf.i2[1] = *(const v4i*)(pw + r * 32 + 16 + half * 8);

#pragma unroll
        for (int t = 0; t < 8; ++t) {
            const bf16* vrow = vbase + (size_t)(t * 16 + r) * S_LEN;
            ABFrag f;
            f.i2[0] = *(const v4i*)(vrow + c0 + half * 8);
            f.i2[1] = *(const v4i*)(vrow + c0 + 16 + half * 8);
            O[t] = __builtin_amdgcn_wmma_f32_16x16x32_bf16(false, pf.v, false, f.v,
                                                           (short)0, O[t], false, false);
        }
    }

    float invl[8];
#pragma unroll
    for (int i = 0; i < 8; ++i) invl[i] = 1.0f / lrow[i];
    const float lam = *lamPtr;

    if (wave == 1) {
#pragma unroll
        for (int t = 0; t < 8; ++t)
#pragma unroll
            for (int i = 0; i < 8; ++i)
                oLds[(i + 8 * half) * 128 + t * 16 + r] = O[t][i] * invl[i];
    }
    __syncthreads();
    if (wave == 0) {
        float out[8][8];
        float ss[8];
#pragma unroll
        for (int i = 0; i < 8; ++i) ss[i] = 0.f;
#pragma unroll
        for (int t = 0; t < 8; ++t)
#pragma unroll
            for (int i = 0; i < 8; ++i) {
                float v = O[t][i] * invl[i] - lam * oLds[(i + 8 * half) * 128 + t * 16 + r];
                out[t][i] = v;
                ss[i] += v * v;
            }
#pragma unroll
        for (int i = 0; i < 8; ++i) {
            for (int d = 1; d < 16; d <<= 1) ss[i] += __shfl_xor(ss[i], d, 32);
            ss[i] = rsqrtf(ss[i] * (1.0f / 128.0f) + EPS_RMS);
        }
#pragma unroll
        for (int t = 0; t < 8; ++t) {
            float w = sub_w[t * 16 + r];
#pragma unroll
            for (int i = 0; i < 8; ++i) {
                int m = i + 8 * half;
                float v = out[t][i] * ss[i] * w;
                ao[((size_t)(b * S_LEN) + q0 + m) * EMB + h * 128 + t * 16 + r] = (bf16)v;
            }
        }
    }
}

// ---------------------------------------------------------------------------
// Host orchestration (graph-capture safe). Workspace ~59.8 MB, layout in bytes:
//   xb@0(8M) wqkvT@8388608(3M) woT@11534336(2M) qkv@13631488(24M)
//   qb@38797312(8M) kb@47185920(2M) vT@49283072(2M) ao@51380224(8M) lam@59768832
// ---------------------------------------------------------------------------
extern "C" void kernel_launch(void* const* d_in, const int* in_sizes, int n_in,
                              void* d_out, int out_size, void* d_ws, size_t ws_size,
                              hipStream_t stream) {
    const float* x    = (const float*)d_in[0];
    const float* cosT = (const float*)d_in[1];
    const float* sinT = (const float*)d_in[2];
    const float* Wq   = (const float*)d_in[3];
    const float* Wk   = (const float*)d_in[4];
    const float* Wv   = (const float*)d_in[5];
    const float* Wo   = (const float*)d_in[6];
    const float* lq1  = (const float*)d_in[7];
    const float* lk1  = (const float*)d_in[8];
    const float* lq2  = (const float*)d_in[9];
    const float* lk2  = (const float*)d_in[10];
    const float* subw = (const float*)d_in[11];

    char* ws = (char*)d_ws;
    bf16*  xb    = (bf16*)(ws + 0);
    bf16*  wqkvT = (bf16*)(ws + 8388608);
    bf16*  woT   = (bf16*)(ws + 11534336);
    float* qkv   = (float*)(ws + 13631488);
    bf16*  qb    = (bf16*)(ws + 38797312);
    bf16*  kb    = (bf16*)(ws + 47185920);
    bf16*  vT    = (bf16*)(ws + 49283072);
    bf16*  ao    = (bf16*)(ws + 51380224);
    float* lam   = (float*)(ws + 59768832);

    cvt_x<<<(M_ROWS * EMB + 255) / 256, 256, 0, stream>>>(x, xb, M_ROWS * EMB);
    pack_wqkvT<<<(NQKV * EMB + 255) / 256, 256, 0, stream>>>(Wq, Wk, Wv, wqkvT);
    pack_woT<<<(EMB * EMB + 255) / 256, 256, 0, stream>>>(Wo, woT);
    lambda_kernel<<<1, 64, 0, stream>>>(lq1, lk1, lq2, lk2, lam);

    // QKV projection: [4096,1024] x [1024,1536], block tile 128x128
    gemm_bf16<<<dim3(NQKV / 128, M_ROWS / 128), 256, 0, stream>>>(
        xb, wqkvT, qkv, M_ROWS, NQKV, EMB, NQKV);

    rope_prep<<<M_ROWS, 256, 0, stream>>>(qkv, cosT, sinT, qb, kb, vT);

    diff_attn<<<BATCH * NUM_HEADS * (S_LEN / 16), 64, 0, stream>>>(
        qb, kb, vT, lam, subw, ao);

    // Output projection: [4096,1024] x [1024,1024] -> f32 d_out
    gemm_bf16<<<dim3(EMB / 128, M_ROWS / 128), 256, 0, stream>>>(
        ao, woT, (float*)d_out, M_ROWS, EMB, EMB, EMB);
}
